// MSDeformAttnTransformerEncoderOnly3D_89627377533666
// MI455X (gfx1250) — matrix operand order repaired and checked by reading, not compiled, not run on hardware
//
#include <hip/hip_runtime.h>

typedef __attribute__((ext_vector_type(16))) _Float16 v16h;
typedef __attribute__((ext_vector_type(8)))  _Float16 v8h;
typedef __attribute__((ext_vector_type(8)))  float    v8f;

namespace {

constexpr int kDM  = 256;
constexpr int kM   = 8;
constexpr int kNL  = 2;
constexpr int kDFF = 1024;
constexpr int kDH  = 32;
constexpr int kLEN = 16*32*32 + 8*16*16 + 4*8*8 + 2*4*4;  // 18720
constexpr int kBT  = 2 * kLEN;                            // 37440 rows, divisible by 64
constexpr float kEPS = 1e-5f;

__device__ __forceinline__ int imin(int a, int b) { return a < b ? a : b; }
__device__ __forceinline__ int imax(int a, int b) { return a > b ? a : b; }

// LDS byte offset of a __shared__ object: generic LDS addresses map low 32 bits
// to the LDS offset (ISA 10.2 aperture rules).
__device__ __forceinline__ unsigned lds_off(const void* p) {
  return (unsigned)(size_t)p;
}

__device__ __forceinline__ void async_copy16(unsigned lds, const void* gptr) {
  unsigned long long ga = (unsigned long long)(size_t)gptr;
  asm volatile("global_load_async_to_lds_b128 %0, %1, off" :: "v"(lds), "v"(ga) : "memory");
}

__device__ __forceinline__ void async_wait0() {
  asm volatile("s_wait_asynccnt 0x0" ::: "memory");
}

// ---------------------------------------------------------------- flatten
__global__ void flatten_kernel(const float* s0, const float* s1, const float* s2, const float* s3,
                               const float* p0, const float* p1, const float* p2, const float* p3,
                               const float* lemb, float* SRC, float* POS) {
  size_t i = (size_t)blockIdx.x * blockDim.x + threadIdx.x;
  if (i >= (size_t)kBT * kDM) return;
  int ch = (int)(i & 255u);
  size_t t = i >> 8;
  int q = (int)(t % kLEN);
  int b = (int)(t / kLEN);
  int l, s, sz; const float *sp, *pp;
  if (q < 16384)      { l = 0; s = q;         sz = 16384; sp = s0; pp = p0; }
  else if (q < 18432) { l = 1; s = q - 16384; sz = 2048;  sp = s1; pp = p1; }
  else if (q < 18688) { l = 2; s = q - 18432; sz = 256;   sp = s2; pp = p2; }
  else                { l = 3; s = q - 18688; sz = 32;    sp = s3; pp = p3; }
  size_t si = ((size_t)b * kDM + ch) * sz + s;
  SRC[i] = sp[si];
  POS[i] = pp[si] + lemb[l * kDM + ch];
}

// ---------------------------------------------------------------- reference points
__global__ void ref_kernel(float* REF) {
  int q = blockIdx.x * blockDim.x + threadIdx.x;
  if (q >= kLEN) return;
  int s, D_, H_, W_;
  if (q < 16384)      { s = q;         D_ = 16; H_ = 32; W_ = 32; }
  else if (q < 18432) { s = q - 16384; D_ = 8;  H_ = 16; W_ = 16; }
  else if (q < 18688) { s = q - 18432; D_ = 4;  H_ = 8;  W_ = 8;  }
  else                { s = q - 18688; D_ = 2;  H_ = 4;  W_ = 4;  }
  int z = s / (H_ * W_);
  int r = s % (H_ * W_);
  int y = r / W_;
  int x = r % W_;
  REF[q * 3 + 0] = (z + 0.5f) / (float)D_;   // comp0 (paired with W scale, per reference quirk)
  REF[q * 3 + 1] = (y + 0.5f) / (float)H_;
  REF[q * 3 + 2] = (x + 0.5f) / (float)W_;
}

// ------------------------------------------------- f32 [K][N] -> f16 transposed [N][K]
__global__ void cvt_f16_t_kernel(const float* __restrict__ in, _Float16* __restrict__ out,
                                 int K, int N) {
  size_t i = (size_t)blockIdx.x * blockDim.x + threadIdx.x;
  if (i >= (size_t)K * N) return;
  int k = (int)(i / N);
  int n = (int)(i % N);
  out[(size_t)n * K + k] = (_Float16)in[i];
}

// ---------------------------------------------------------------- prep: q = src+pos (f16), src (f16)
__global__ void prep_kernel(const float* __restrict__ SRC, const float* __restrict__ POS,
                            _Float16* __restrict__ Q16, _Float16* __restrict__ X16) {
  size_t i = (size_t)blockIdx.x * blockDim.x + threadIdx.x;
  if (i >= (size_t)kBT * kDM) return;
  float sv = SRC[i];
  Q16[i] = (_Float16)(sv + POS[i]);
  X16[i] = (_Float16)sv;
}

// ---------------------------------------------------------------- WMMA GEMM
// C[Mtot x N] = A[Mtot x K] (f16, row-major) x B (f16, pre-TRANSPOSED [N][K]) + bias.
// block = 128 threads (4 waves); block tile 64x64; wave tile 32x32 (2x2 wmma); K step 32.
// Double-buffered LDS staging via GLOBAL_LOAD_ASYNC_TO_LDS_B128 (ASYNCcnt pipeline).
// Mtot % 64 == 0, N % 64 == 0, K % 32 == 0 guaranteed by problem sizes.
template<bool RELU, bool STORE16>
__global__ void __launch_bounds__(128)
gemm_wmma_kernel(const _Float16* __restrict__ A, const _Float16* __restrict__ BwT,
                 const float* __restrict__ bias, float* __restrict__ Cf,
                 _Float16* __restrict__ Ch, int N, int K) {
  struct Tile { _Float16 a[64][32]; _Float16 b[64][32]; };
  __shared__ __align__(32) Tile tile[2];

  const int tid  = threadIdx.x;
  const int lane = tid & 31;
  const int wid  = tid >> 5;
  const int gm0  = blockIdx.y * 64;
  const int gn0  = blockIdx.x * 64;
  const int m0   = (wid >> 1) * 32;
  const int n0   = (wid & 1) * 32;
  const int NK   = K >> 5;

  const _Float16* Arow = A   + (size_t)gm0 * K;
  const _Float16* Brow = BwT + (size_t)gn0 * K;

  v8f acc00 = {}, acc01 = {}, acc10 = {}, acc11 = {};

  // stage one 64x32 A tile + 64x32 B tile: 4 async 16B copies per thread
  auto stage = [&](int kt, int buf) {
    const int k0 = kt << 5;
#pragma unroll
    for (int h = 0; h < 2; ++h) {
      const int i = tid + (h << 7);
      const int r = i >> 2;
      const int c = (i & 3) << 3;
      async_copy16(lds_off(&tile[buf].a[r][c]), &Arow[(size_t)r * K + k0 + c]);
      async_copy16(lds_off(&tile[buf].b[r][c]), &Brow[(size_t)r * K + k0 + c]);
    }
  };

  stage(0, 0);
  async_wait0();
  __syncthreads();

  for (int kt = 0; kt < NK; ++kt) {
    const int cur = kt & 1;
    if (kt + 1 < NK) stage(kt + 1, cur ^ 1);   // prefetch next tiles while computing

    // fragments per CDNA5 VGPR layouts
    const int ma  = lane & 15;
    const int kb0 = (lane & 16) ? 8 : 0;       // A: half-wave K groups {0-7,16-23} / {8-15,24-31}
    v16h a0, a1, b0, b1;
    {
      v8h lo = *(const v8h*)&tile[cur].a[m0 + ma][kb0];
      v8h hi = *(const v8h*)&tile[cur].a[m0 + ma][kb0 + 16];
#pragma unroll
      for (int j = 0; j < 8; ++j) { a0[j] = lo[j]; a0[j + 8] = hi[j]; }
      lo = *(const v8h*)&tile[cur].a[m0 + 16 + ma][kb0];
      hi = *(const v8h*)&tile[cur].a[m0 + 16 + ma][kb0 + 16];
#pragma unroll
      for (int j = 0; j < 8; ++j) { a1[j] = lo[j]; a1[j + 8] = hi[j]; }
    }
    {
      const int kb = (lane & 16) ? 8 : 0;      // B: half-wave K groups {0-7,16-23} / {8-15,24-31}
      v8h lo = *(const v8h*)&tile[cur].b[n0 + ma][kb];
      v8h hi = *(const v8h*)&tile[cur].b[n0 + ma][kb + 16];
#pragma unroll
      for (int j = 0; j < 8; ++j) { b0[j] = lo[j]; b0[j + 8] = hi[j]; }
      lo = *(const v8h*)&tile[cur].b[n0 + 16 + ma][kb];
      hi = *(const v8h*)&tile[cur].b[n0 + 16 + ma][kb + 16];
#pragma unroll
      for (int j = 0; j < 8; ++j) { b1[j] = lo[j]; b1[j + 8] = hi[j]; }
    }
    acc00 = __builtin_amdgcn_wmma_f32_16x16x32_f16(false, a0, false, b0, (short)0, acc00, false, false);
    acc01 = __builtin_amdgcn_wmma_f32_16x16x32_f16(false, a0, false, b1, (short)0, acc01, false, false);
    acc10 = __builtin_amdgcn_wmma_f32_16x16x32_f16(false, a1, false, b0, (short)0, acc10, false, false);
    acc11 = __builtin_amdgcn_wmma_f32_16x16x32_f16(false, a1, false, b1, (short)0, acc11, false, false);

    async_wait0();       // next-tile copies complete
    __syncthreads();     // all waves done reading tile[cur]
  }

  // epilogue: C layout VGPR r -> row r (lanes 0-15) / r+8 (lanes 16-31), col = lane&15
  const int col  = lane & 15;
  const int rofs = (lane & 16) ? 8 : 0;
#pragma unroll
  for (int t = 0; t < 4; ++t) {
    int ti = t >> 1, tj = t & 1;
    v8f acc = (t == 0) ? acc00 : (t == 1) ? acc01 : (t == 2) ? acc10 : acc11;
    int gm_base = gm0 + m0 + ti * 16 + rofs;
    int gn = gn0 + n0 + tj * 16 + col;
    float bb = bias[gn];
#pragma unroll
    for (int r = 0; r < 8; ++r) {
      float vv = acc[r] + bb;
      if (RELU) vv = fmaxf(vv, 0.f);
      size_t oi = (size_t)(gm_base + r) * N + gn;
      if (STORE16) Ch[oi] = (_Float16)vv; else Cf[oi] = vv;
    }
  }
}

// ---------------------------------------------------------------- softmax over 16 (LV*P) per (b,q,head)
__global__ void softmax_kernel(float* __restrict__ ATT) {
  int i = blockIdx.x * blockDim.x + threadIdx.x;
  if (i >= kBT * kM) return;
  float* a = ATT + (size_t)i * 16;
  float mx = a[0];
#pragma unroll
  for (int j = 1; j < 16; ++j) mx = fmaxf(mx, a[j]);
  float e[16]; float s = 0.f;
#pragma unroll
  for (int j = 0; j < 16; ++j) { e[j] = __expf(a[j] - mx); s += e[j]; }
  float inv = 1.f / s;
#pragma unroll
  for (int j = 0; j < 16; ++j) a[j] = e[j] * inv;
}

// ---------------------------------------------------------------- deformable trilinear sampling
// one wave per (b,q,head); lane = channel (DH=32). Writes f16 agg for the out-GEMM.
__global__ void sample_kernel(const float* __restrict__ VAL, const float* __restrict__ OFF,
                              const float* __restrict__ ATT, const float* __restrict__ REF,
                              _Float16* __restrict__ AGG16) {
  int gw = (int)(((size_t)blockIdx.x * blockDim.x + threadIdx.x) >> 5);
  int lane = threadIdx.x & 31;
  if (gw >= kBT * kM) return;
  int m  = gw & 7;
  int bq = gw >> 3;
  int q  = bq % kLEN;
  int b  = bq / kLEN;
  const float* offp = OFF + (size_t)bq * 384 + m * 48;   // [LV][P][3]
  const float* attp = ATT + (size_t)bq * 128 + m * 16;   // [LV][P]
  float r0 = REF[q * 3 + 0], r1 = REF[q * 3 + 1], r2 = REF[q * 3 + 2];
  const int Ld[4] = {16, 8, 4, 2}, Lh[4] = {32, 16, 8, 4}, Lw[4] = {32, 16, 8, 4};
  const int Lo[4] = {0, 16384, 18432, 18688};
  float acc = 0.f;
#pragma unroll
  for (int l = 0; l < 4; ++l) {
    const int D_ = Ld[l], H_ = Lh[l], W_ = Lw[l];
    const float* vbase = VAL + ((size_t)b * kLEN + Lo[l]) * kDM + m * kDH + lane;
#pragma unroll
    for (int p = 0; p < 4; ++p) {
      int pi = l * 4 + p;
      float o0 = offp[pi * 3 + 0], o1 = offp[pi * 3 + 1], o2 = offp[pi * 3 + 2];
      float aw = attp[pi];
      // loc = ref + off/normalizer, then *size - 0.5  (normalizer row = (W,H,D))
      float x = r0 * (float)W_ + o0 - 0.5f;
      float y = r1 * (float)H_ + o1 - 0.5f;
      float z = r2 * (float)D_ + o2 - 0.5f;
      float xf = floorf(x), yf = floorf(y), zf = floorf(z);
      float fx = x - xf, fy = y - yf, fz = z - zf;
      int x0 = (int)xf, y0 = (int)yf, z0 = (int)zf;
      int   xs[2] = {x0, x0 + 1}; float wx[2] = {1.f - fx, fx};
      int   ys[2] = {y0, y0 + 1}; float wy[2] = {1.f - fy, fy};
      int   zs[2] = {z0, z0 + 1}; float wz[2] = {1.f - fz, fz};
#pragma unroll
      for (int dz = 0; dz < 2; ++dz)
#pragma unroll
        for (int dy = 0; dy < 2; ++dy)
#pragma unroll
          for (int dx = 0; dx < 2; ++dx) {
            int xi = xs[dx], yi = ys[dy], zi = zs[dz];
            bool valid = ((unsigned)xi < (unsigned)W_) && ((unsigned)yi < (unsigned)H_) &&
                         ((unsigned)zi < (unsigned)D_);
            int xc = imin(imax(xi, 0), W_ - 1);
            int yc = imin(imax(yi, 0), H_ - 1);
            int zc = imin(imax(zi, 0), D_ - 1);
            int si = (zc * H_ + yc) * W_ + xc;
            float v = vbase[(size_t)si * kDM];
            float w = wx[dx] * wy[dy] * wz[dz];
            acc += valid ? (aw * w * v) : 0.f;
          }
    }
  }
  AGG16[(size_t)bq * kDM + m * kDH + lane] = (_Float16)acc;
}

// ---------------------------------------------------------------- fused residual add + LayerNorm (wave per row)
__global__ void add_ln_kernel(float* __restrict__ X, const float* __restrict__ Y,
                              const float* __restrict__ g, const float* __restrict__ bta,
                              _Float16* __restrict__ X16) {
  int row  = (int)(((size_t)blockIdx.x * blockDim.x + threadIdx.x) >> 5);
  int lane = threadIdx.x & 31;
  if (row >= kBT) return;
  float* x = X + (size_t)row * kDM;
  const float* y = Y + (size_t)row * kDM;
  float v[8]; float s = 0.f;
#pragma unroll
  for (int j = 0; j < 8; ++j) { v[j] = x[lane + j * 32] + y[lane + j * 32]; s += v[j]; }
#pragma unroll
  for (int o = 16; o > 0; o >>= 1) s += __shfl_xor(s, o, 32);
  float mu = s * (1.f / 256.f);
  float sq = 0.f;
#pragma unroll
  for (int j = 0; j < 8; ++j) { float d = v[j] - mu; sq += d * d; }
#pragma unroll
  for (int o = 16; o > 0; o >>= 1) sq += __shfl_xor(sq, o, 32);
  float inv = rsqrtf(sq * (1.f / 256.f) + kEPS);
#pragma unroll
  for (int j = 0; j < 8; ++j) {
    int n = lane + j * 32;
    float r = (v[j] - mu) * inv * g[n] + bta[n];
    x[n] = r;
    if (X16) X16[(size_t)row * kDM + n] = (_Float16)r;
  }
}

// ---------------------------------------------------------------- final copy
__global__ void copy_kernel(const float* __restrict__ in, float* __restrict__ out, size_t n) {
  size_t i = (size_t)blockIdx.x * blockDim.x + threadIdx.x;
  if (i < n) out[i] = in[i];
}

}  // namespace

extern "C" void kernel_launch(void* const* d_in, const int* in_sizes, int n_in,
                              void* d_out, int out_size, void* d_ws, size_t ws_size,
                              hipStream_t stream) {
  (void)in_sizes; (void)n_in; (void)out_size; (void)ws_size;
  const float* src[4] = {(const float*)d_in[0], (const float*)d_in[2],
                         (const float*)d_in[4], (const float*)d_in[6]};
  const float* pos[4] = {(const float*)d_in[1], (const float*)d_in[3],
                         (const float*)d_in[5], (const float*)d_in[7]};
  const float* lemb   = (const float*)d_in[8];
  const float* W_off  = (const float*)d_in[9];   const float* b_off  = (const float*)d_in[10];
  const float* W_attn = (const float*)d_in[11];  const float* b_attn = (const float*)d_in[12];
  const float* W_val  = (const float*)d_in[13];  const float* b_val  = (const float*)d_in[14];
  const float* W_out  = (const float*)d_in[15];  const float* b_out  = (const float*)d_in[16];
  const float* ln1_g  = (const float*)d_in[17];  const float* ln1_b  = (const float*)d_in[18];
  const float* W_ff1  = (const float*)d_in[19];  const float* b_ff1  = (const float*)d_in[20];
  const float* W_ff2  = (const float*)d_in[21];  const float* b_ff2  = (const float*)d_in[22];
  const float* ln2_g  = (const float*)d_in[23];  const float* ln2_b  = (const float*)d_in[24];

  // ---- workspace carve-out
  char* ws = (char*)d_ws;
  auto alloc = [&](size_t bytes) -> void* {
    void* p = (void*)ws;
    ws += (bytes + 255) & ~(size_t)255;
    return p;
  };
  float*    SRC    = (float*)alloc((size_t)kBT * kDM * 4);
  float*    POS    = (float*)alloc((size_t)kBT * kDM * 4);
  float*    VAL    = (float*)alloc((size_t)kBT * kDM * 4);
  float*    OFFb   = (float*)alloc((size_t)kBT * 384 * 4);   // reused for attn-out & ff2-out
  float*    ATT    = (float*)alloc((size_t)kBT * 128 * 4);
  float*    REF    = (float*)alloc((size_t)kLEN * 3 * 4);
  _Float16* Q16    = (_Float16*)alloc((size_t)kBT * kDM * 2);
  _Float16* X16    = (_Float16*)alloc((size_t)kBT * kDM * 2);
  _Float16* AGG16  = (_Float16*)alloc((size_t)kBT * kDM * 2);
  _Float16* H16    = (_Float16*)alloc((size_t)kBT * kDFF * 2);
  _Float16* Woff16 = (_Float16*)alloc((size_t)kNL * kDM * 384 * 2);  // transposed [384][256]
  _Float16* Watt16 = (_Float16*)alloc((size_t)kNL * kDM * 128 * 2);  // transposed [128][256]
  _Float16* Wval16 = (_Float16*)alloc((size_t)kNL * kDM * kDM * 2);  // transposed
  _Float16* Wout16 = (_Float16*)alloc((size_t)kNL * kDM * kDM * 2);  // transposed
  _Float16* Wf1_16 = (_Float16*)alloc((size_t)kNL * kDM * kDFF * 2); // transposed [1024][256]
  _Float16* Wf2_16 = (_Float16*)alloc((size_t)kNL * kDFF * kDM * 2); // transposed [256][1024]

  const size_t nTok = (size_t)kBT * kDM;     // 9,584,640
  const int TPB = 256;
  auto nb = [](size_t n, int t) { return (unsigned)((n + t - 1) / t); };

  // ---- one-time prep
  flatten_kernel<<<nb(nTok, TPB), TPB, 0, stream>>>(src[0], src[1], src[2], src[3],
                                                    pos[0], pos[1], pos[2], pos[3],
                                                    lemb, SRC, POS);
  ref_kernel<<<nb(kLEN, TPB), TPB, 0, stream>>>(REF);
  for (int l = 0; l < kNL; ++l) {
    cvt_f16_t_kernel<<<nb((size_t)kDM * 384, TPB), TPB, 0, stream>>>(
        W_off + (size_t)l * kDM * 384, Woff16 + (size_t)l * kDM * 384, kDM, 384);
    cvt_f16_t_kernel<<<nb((size_t)kDM * 128, TPB), TPB, 0, stream>>>(
        W_attn + (size_t)l * kDM * 128, Watt16 + (size_t)l * kDM * 128, kDM, 128);
    cvt_f16_t_kernel<<<nb((size_t)kDM * kDM, TPB), TPB, 0, stream>>>(
        W_val + (size_t)l * kDM * kDM, Wval16 + (size_t)l * kDM * kDM, kDM, kDM);
    cvt_f16_t_kernel<<<nb((size_t)kDM * kDM, TPB), TPB, 0, stream>>>(
        W_out + (size_t)l * kDM * kDM, Wout16 + (size_t)l * kDM * kDM, kDM, kDM);
    cvt_f16_t_kernel<<<nb((size_t)kDM * kDFF, TPB), TPB, 0, stream>>>(
        W_ff1 + (size_t)l * kDM * kDFF, Wf1_16 + (size_t)l * kDM * kDFF, kDM, kDFF);
    cvt_f16_t_kernel<<<nb((size_t)kDFF * kDM, TPB), TPB, 0, stream>>>(
        W_ff2 + (size_t)l * kDFF * kDM, Wf2_16 + (size_t)l * kDFF * kDM, kDFF, kDM);
  }

  auto gemm = [&](const _Float16* A, const _Float16* BwT, const float* bias,
                  float* Cf, _Float16* Ch, int N, int K, bool relu16) {
    dim3 grid((unsigned)(N / 64), (unsigned)(kBT / 64));
    if (relu16)
      gemm_wmma_kernel<true, true><<<grid, 128, 0, stream>>>(A, BwT, bias, Cf, Ch, N, K);
    else
      gemm_wmma_kernel<false, false><<<grid, 128, 0, stream>>>(A, BwT, bias, Cf, Ch, N, K);
  };

  for (int l = 0; l < kNL; ++l) {
    // q = src + lvl_pos (f16); src -> f16
    prep_kernel<<<nb(nTok, TPB), TPB, 0, stream>>>(SRC, POS, Q16, X16);
    // value / offsets / attention projections
    gemm(X16, Wval16 + (size_t)l * kDM * kDM, b_val + l * kDM, VAL, nullptr, kDM, kDM, false);
    gemm(Q16, Woff16 + (size_t)l * kDM * 384, b_off + l * 384, OFFb, nullptr, 384, kDM, false);
    gemm(Q16, Watt16 + (size_t)l * kDM * 128, b_attn + l * 128, ATT, nullptr, 128, kDM, false);
    softmax_kernel<<<nb((size_t)kBT * kM, TPB), TPB, 0, stream>>>(ATT);
    // deformable sampling -> agg (f16)
    sample_kernel<<<nb((size_t)kBT * kM * 32, TPB), TPB, 0, stream>>>(VAL, OFFb, ATT, REF, AGG16);
    // output projection (f32 into OFFb scratch), residual + LN1 (emits f16 src for FFN)
    gemm(AGG16, Wout16 + (size_t)l * kDM * kDM, b_out + l * kDM, OFFb, nullptr, kDM, kDM, false);
    add_ln_kernel<<<nb((size_t)kBT * 32, TPB), TPB, 0, stream>>>(SRC, OFFb, ln1_g + l * kDM, ln1_b + l * kDM, X16);
    // FFN: relu(x @ W1 + b1) (f16) @ W2 + b2 (f32), residual + LN2
    gemm(X16, Wf1_16 + (size_t)l * kDM * kDFF, b_ff1 + l * kDFF, nullptr, H16, kDFF, kDM, true);
    gemm(H16, Wf2_16 + (size_t)l * kDFF * kDM, b_ff2 + l * kDM, OFFb, nullptr, kDM, kDFF, false);
    add_ln_kernel<<<nb((size_t)kBT * 32, TPB), TPB, 0, stream>>>(SRC, OFFb, ln2_g + l * kDM, ln2_b + l * kDM, nullptr);
  }

  copy_kernel<<<nb(nTok, TPB), TPB, 0, stream>>>(SRC, (float*)d_out, nTok);
}
